// Season_Model_block_86053964743195
// MI455X (gfx1250) — matrix operand order repaired and checked by reading, not compiled, not run on hardware
//
#include <hip/hip_runtime.h>
#include <math.h>

typedef __attribute__((ext_vector_type(16))) _Float16 v16h;
typedef __attribute__((ext_vector_type(8)))  _Float16 v8h;
typedef __attribute__((ext_vector_type(8)))  float    v8f;

#define TILE_M 128
#define TILE_N 128
#define TILE_K 32
#define APAD   40   // f16 per LDS row: 32 + 8 pad -> 80B stride (16B multiple)

// issue one per-lane 16B async copy global -> LDS (ASYNCcnt++)
__device__ __forceinline__ void async_copy_b128(unsigned lds, const _Float16* g) {
  asm volatile("global_load_async_to_lds_b128 %0, %1, off"
               :: "v"(lds), "v"(g) : "memory");
}
#define ASYNC_WAIT(n) asm volatile("s_wait_asynccnt " #n ::: "memory")

__device__ __forceinline__ unsigned lds_addr(const void* p) {
  return (unsigned)(uintptr_t)p;   // low 32 bits of flat shared addr == LDS byte offset
}

// 2x4 wave tile of 16x16x32 WMMAs out of LDS tiles As[row][k], Bt[col][k]
__device__ __forceinline__ void wmma_tiles(const _Float16* As, const _Float16* Bt,
                                           v8f acc[2][4], int lane, int waveM, int waveN)
{
  #pragma unroll
  for (int mi = 0; mi < 2; ++mi) {
    // A frag: lanes 0-15 K 0-7/16-23, lanes 16-31 K 8-15/24-31
    int arow  = waveM * 32 + mi * 16 + (lane & 15);
    int khalf = (lane >> 4) * 8;
    const _Float16* ap = &As[arow * APAD + khalf];
    v8h alo = *(const v8h*)ap;
    v8h ahi = *(const v8h*)(ap + 16);
    v16h af;
    #pragma unroll
    for (int j = 0; j < 8; ++j) { af[j] = alo[j]; af[j + 8] = ahi[j]; }

    #pragma unroll
    for (int ni = 0; ni < 4; ++ni) {
      // B frag: K-major per column; lanes 0-15 K 0-7/16-23, lanes 16-31 K 8-15/24-31
      int bcol = waveN * 64 + ni * 16 + (lane & 15);
      int koff = (lane >> 4) * 8;
      const _Float16* bp = &Bt[bcol * APAD + koff];
      v8h blo = *(const v8h*)bp;
      v8h bhi = *(const v8h*)(bp + 16);
      v16h bf;
      #pragma unroll
      for (int j = 0; j < 8; ++j) { bf[j] = blo[j]; bf[j + 8] = bhi[j]; }

      acc[mi][ni] = __builtin_amdgcn_wmma_f32_16x16x32_f16(
          false, af, false, bf, (short)0, acc[mi][ni], false, false);
    }
  }
}

// ---------------------------------------------------------------------------
// C[M,N] = A[M,K] @ Wt[N,K]^T (+bias). A, Wt are f16; C is f16 or f32.
// 256 threads = 8 waves (4 along M x 2 along N); block tile 128x128, K step 32.
// Full K tiles use double-buffered global_load_async_to_lds_b128; the K tail
// (K % 32) uses a guarded scalar fill so the hot loop has zero branches.
// ---------------------------------------------------------------------------
template <bool OUT_HALF>
__global__ __launch_bounds__(256) void gemm_wmma_f16(
    const _Float16* __restrict__ A,   // [M,K]
    const _Float16* __restrict__ Bt,  // [N,K] pre-transposed weights
    const float* __restrict__ bias,
    void* __restrict__ Cout,
    int M, int N, int K)
{
  __shared__ _Float16 As[2][TILE_M * APAD];
  __shared__ _Float16 Bs[2][TILE_N * APAD];

  const int tid   = threadIdx.x;
  const int lane  = tid & 31;
  const int wave  = tid >> 5;
  const int waveM = wave & 3;
  const int waveN = wave >> 2;
  const int mBase = blockIdx.x * TILE_M;
  const int nBase = blockIdx.y * TILE_N;

  v8f acc[2][4] = {};

  const int kFull = K / TILE_K;
  const int kRem  = K - kFull * TILE_K;

  // issue all async copies for one K tile (exactly 4 per thread -> 4 per wave)
  auto issue = [&](int buf, int k0) {
    #pragma unroll
    for (int i = 0; i < 2; ++i) {
      int c   = tid + i * 256;          // 0..511
      int row = c >> 2;                 // 0..127
      int kq  = (c & 3) * 8;            // 0,8,16,24
      async_copy_b128(lds_addr(&As[buf][row * APAD + kq]),
                      A + (size_t)(mBase + row) * K + k0 + kq);
    }
    #pragma unroll
    for (int i = 0; i < 2; ++i) {
      int c   = tid + i * 256;
      int col = c >> 2;
      int kq  = (c & 3) * 8;
      // clamp OOB columns to a safe address: garbage only feeds output
      // columns >= N, which are never stored.
      int colc = nBase + col; if (colc > N - 1) colc = N - 1;
      async_copy_b128(lds_addr(&Bs[buf][col * APAD + kq]),
                      Bt + (size_t)colc * K + k0 + kq);
    }
  };

  if (kFull > 0) issue(0, 0);

  for (int kt = 0; kt < kFull; ++kt) {
    if (kt + 1 < kFull) {
      issue((kt + 1) & 1, (kt + 1) * TILE_K);
      ASYNC_WAIT(0x4);                  // current tile's 4 copies done; next in flight
    } else {
      ASYNC_WAIT(0x0);
    }
    __syncthreads();
    wmma_tiles(As[kt & 1], Bs[kt & 1], acc, lane, waveM, waveN);
    __syncthreads();
  }

  if (kRem) {
    const int k0  = kFull * TILE_K;
    const int buf = kFull & 1;
    for (int i = tid; i < TILE_M * TILE_K; i += 256) {
      int row = i >> 5, k = i & 31;
      _Float16 vA = (_Float16)0.f;
      if (k0 + k < K) vA = A[(size_t)(mBase + row) * K + k0 + k];
      As[buf][row * APAD + k] = vA;
    }
    for (int i = tid; i < TILE_N * TILE_K; i += 256) {
      int col = i >> 5, k = i & 31;
      _Float16 vB = (_Float16)0.f;
      if ((k0 + k < K) && (nBase + col < N))
        vB = Bt[(size_t)(nBase + col) * K + k0 + k];
      Bs[buf][col * APAD + k] = vB;
    }
    __syncthreads();
    wmma_tiles(As[buf], Bs[buf], acc, lane, waveM, waveN);
  }

  // store (C/D layout: lanes 0-15 N=lane M=v; lanes 16-31 N=lane-16 M=8+v)
  #pragma unroll
  for (int mi = 0; mi < 2; ++mi) {
    #pragma unroll
    for (int ni = 0; ni < 4; ++ni) {
      int n = nBase + waveN * 64 + ni * 16 + (lane & 15);
      if (n < N) {
        float bv = bias ? bias[n] : 0.0f;
        int m0 = mBase + waveM * 32 + mi * 16 + (lane >> 4) * 8;
        #pragma unroll
        for (int v = 0; v < 8; ++v) {
          float r = acc[mi][ni][v] + bv;
          if constexpr (OUT_HALF)
            ((_Float16*)Cout)[(size_t)(m0 + v) * N + n] = (_Float16)r;
          else
            ((float*)Cout)[(size_t)(m0 + v) * N + n] = r;
        }
      }
    }
  }
}

// ---------------------------------------------------------------------------
// W[K,N] f32 -> Wt[N,K] f16 (pre-transpose so GEMM B loads are contiguous)
// ---------------------------------------------------------------------------
__global__ __launch_bounds__(256) void convert_transpose_w(
    const float* __restrict__ W, _Float16* __restrict__ Wt, int K, int N)
{
  int idx = blockIdx.x * 256 + threadIdx.x;
  if (idx >= K * N) return;
  int n = idx / K, k = idx % K;
  Wt[idx] = (_Float16)W[(size_t)k * N + n];
}

// ---------------------------------------------------------------------------
// Pack [B,L,H,E] f32 -> [(B*H*E), L] f16
// ---------------------------------------------------------------------------
__global__ __launch_bounds__(256) void pack_qkv(const float* __restrict__ x,
                                                _Float16* __restrict__ P)
{
  const int NL = 336, HH = 8, EE = 64;
  int idx = blockIdx.x * 256 + threadIdx.x;
  if (idx >= 16384 * NL) return;
  int r = idx / NL, l = idx % NL;
  int b = r >> 9;
  int he = r & 511;
  int h = he >> 6, e = he & 63;
  P[idx] = (_Float16)x[(((size_t)b * NL + l) * HH + h) * EE + e];
}

// ---------------------------------------------------------------------------
// Per-(b,h): scores = tq^T tk / T, softmax over x axis, qkv = attn . tv.
// ---------------------------------------------------------------------------
__global__ __launch_bounds__(1024) void attn_kernel(
    const float* __restrict__ Tq, const float* __restrict__ Tk,
    const float* __restrict__ Tv, float* __restrict__ Out)
{
  __shared__ float tq[64][32];
  __shared__ float tk[64][32];
  __shared__ float tv[64][32];
  __shared__ float at[32][33];

  const int bh = blockIdx.x;
  const int tx = threadIdx.x;          // x (0..31)
  const int ty = threadIdx.y;          // y (0..31)

  #pragma unroll
  for (int i = 0; i < 2; ++i) {
    int e = ty + i * 32;
    size_t off = ((size_t)bh * 64 + e) * 32 + tx;
    tq[e][tx] = Tq[off];
    tk[e][tx] = Tk[off];
    tv[e][tx] = Tv[off];
  }
  __syncthreads();

  float s = 0.0f;
  #pragma unroll 8
  for (int e = 0; e < 64; ++e) s += tq[e][tx] * tk[e][ty];
  s *= (1.0f / 32.0f);

  float mx = s;
  #pragma unroll
  for (int off = 16; off > 0; off >>= 1) mx = fmaxf(mx, __shfl_xor(mx, off, 32));
  float ex = __expf(s - mx);
  float sm = ex;
  #pragma unroll
  for (int off = 16; off > 0; off >>= 1) sm += __shfl_xor(sm, off, 32);
  at[tx][ty] = ex / sm;
  __syncthreads();

  #pragma unroll
  for (int i = 0; i < 2; ++i) {
    int e = ty + i * 32;
    float acc = 0.0f;
    #pragma unroll 8
    for (int y = 0; y < 32; ++y) acc += at[tx][y] * tv[e][y];
    Out[((size_t)bh * 64 + e) * 32 + tx] = acc;
  }
}

// ---------------------------------------------------------------------------
// out[b,d,l] = sum_t thetas[b*512+d, t] * Smat[t, l]; Smat built on the fly.
// ---------------------------------------------------------------------------
__global__ __launch_bounds__(256) void season_kernel(
    const float* __restrict__ Th, const float* __restrict__ F,
    float* __restrict__ Out)
{
  __shared__ float Sm[32][113];
  __shared__ float Tt[128][33];

  const int b   = blockIdx.x;
  const int d0  = blockIdx.y * 128;
  const int l0  = blockIdx.z * 112;
  const int tid = threadIdx.x;
  const float w = 6.283185307179586f / 336.0f;

  for (int i = tid; i < 32 * 112; i += 256) {
    int t = i / 112, l = i % 112;
    float f = F[t];
    float harm = (float)(t < 16 ? t : t - 16);
    float arg = w * (float)(l0 + l) * f * harm;
    Sm[t][l] = (t < 16) ? __cosf(arg) : __sinf(arg);
  }
  for (int i = tid; i < 128 * 32; i += 256) {
    int d = i >> 5, t = i & 31;
    Tt[d][t] = Th[((size_t)b * 512 + d0 + d) * 32 + t];
  }
  __syncthreads();

  for (int i = tid; i < 128 * 112; i += 256) {
    int d = i / 112, l = i % 112;
    float acc = 0.0f;
    #pragma unroll
    for (int t = 0; t < 32; ++t) acc += Tt[d][t] * Sm[t][l];
    Out[((size_t)b * 512 + d0 + d) * 336 + l0 + l] = acc;
  }
}

// ---------------------------------------------------------------------------
extern "C" void kernel_launch(void* const* d_in, const int* in_sizes, int n_in,
                              void* d_out, int out_size, void* d_ws, size_t ws_size,
                              hipStream_t stream)
{
  (void)in_sizes; (void)n_in; (void)out_size; (void)ws_size;

  const float* q = (const float*)d_in[0];
  const float* k = (const float*)d_in[1];
  const float* v = (const float*)d_in[2];
  // d_in[3] = mask (unused)
  const float* W_in[3]  = {(const float*)d_in[4],  (const float*)d_in[9],  (const float*)d_in[14]};
  const float* b_in[3]  = {(const float*)d_in[5],  (const float*)d_in[10], (const float*)d_in[15]};
  const float* W_mid[3] = {(const float*)d_in[6],  (const float*)d_in[11], (const float*)d_in[16]};
  const float* b_mid[3] = {(const float*)d_in[7],  (const float*)d_in[12], (const float*)d_in[17]};
  const float* W_out[3] = {(const float*)d_in[8],  (const float*)d_in[13], (const float*)d_in[18]};
  const float* F_i = (const float*)d_in[19];
  const float* X[3] = {q, k, v};

  const int M = 16384, U = 1008, Lq = 336, T = 32;

  char* base = (char*)d_ws;
  _Float16* P16 = (_Float16*)base;  base += (size_t)M * Lq * 2;     // 11 MB
  _Float16* Wt  = (_Float16*)base;  base += (size_t)U * U * 2;      // 2 MB (reused)
  _Float16* H0  = (_Float16*)base;  base += (size_t)M * U * 2;      // 33 MB
  _Float16* H1  = (_Float16*)base;  base += (size_t)M * U * 2;      // 33 MB
  float* Th[3];
  Th[0] = (float*)base;             base += (size_t)M * T * 4;
  Th[1] = (float*)base;             base += (size_t)M * T * 4;
  Th[2] = (float*)base;             base += (size_t)M * T * 4;
  float* Qk = (float*)base;

  dim3 blk(256);
  dim3 gPack((M * Lq + 255) / 256);
  dim3 gU(M / 128, (U + 127) / 128);
  dim3 gT(M / 128, 1);
  dim3 gWin((Lq * U + 255) / 256);
  dim3 gWmid((U * U + 255) / 256);
  dim3 gWout((U * T + 255) / 256);

  for (int s = 0; s < 3; ++s) {
    pack_qkv<<<gPack, blk, 0, stream>>>(X[s], P16);

    convert_transpose_w<<<gWin, blk, 0, stream>>>(W_in[s], Wt, Lq, U);
    gemm_wmma_f16<true><<<gU, blk, 0, stream>>>(P16, Wt, b_in[s], H0, M, U, Lq);

    convert_transpose_w<<<gWmid, blk, 0, stream>>>(W_mid[s], Wt, U, U);
    gemm_wmma_f16<true><<<gU, blk, 0, stream>>>(H0, Wt, b_mid[s], H1, M, U, U);

    convert_transpose_w<<<gWmid, blk, 0, stream>>>(W_mid[s] + (size_t)U * U, Wt, U, U);
    gemm_wmma_f16<true><<<gU, blk, 0, stream>>>(H1, Wt, b_mid[s] + U, H0, M, U, U);

    convert_transpose_w<<<gWmid, blk, 0, stream>>>(W_mid[s] + 2 * (size_t)U * U, Wt, U, U);
    gemm_wmma_f16<true><<<gU, blk, 0, stream>>>(H0, Wt, b_mid[s] + 2 * U, H1, M, U, U);

    convert_transpose_w<<<gWout, blk, 0, stream>>>(W_out[s], Wt, U, T);
    gemm_wmma_f16<false><<<gT, blk, 0, stream>>>(H1, Wt, nullptr, Th[s], M, T, U);
  }

  attn_kernel<<<dim3(256), dim3(32, 32), 0, stream>>>(Th[0], Th[1], Th[2], Qk);
  season_kernel<<<dim3(32, 4, 3), dim3(256), 0, stream>>>(Qk, F_i, (float*)d_out);
}
